// DisjGNN_15693810500067
// MI455X (gfx1250) — matrix operand order repaired and checked by reading, not compiled, not run on hardware
//
#include <hip/hip_runtime.h>
#include <hip/hip_bf16.h>

typedef _Float16 v16h __attribute__((ext_vector_type(16)));
typedef float    v8f  __attribute__((ext_vector_type(8)));

#define WPB 8     // waves per block (wave32)
#define TPB 256   // threads per block

__global__ __launch_bounds__(TPB) void disj_gnn_fused(
    const float* __restrict__ nf,  const int* __restrict__ ei,
    const float* __restrict__ gW1, const float* __restrict__ gB1,
    const float* __restrict__ gW2, const float* __restrict__ gB2,
    const float* __restrict__ gW3, const float* __restrict__ gB3,
    const float* __restrict__ gW4, const float* __restrict__ gB4,
    const float* __restrict__ gW5, const float* __restrict__ gB5,
    const float* __restrict__ gW6, const float* __restrict__ gB6,
    float* __restrict__ out_attr, float* __restrict__ out_cos,
    int E, int n_tiles)
{
    // W1 in WMMA B-fragment order (lane-major, 16 halfs per lane per fragment)
    __shared__ __align__(32) _Float16 sW1f[4 * 4 * 32 * 16]; // [kchunk][ntile][lane][e]
    // Per-wave activation staging (row-major f16)
    __shared__ __align__(16) _Float16 sH1[WPB][16 * 64];
    __shared__ __align__(16) _Float16 sH2[WPB][16 * 32];
    __shared__ __align__(16) _Float16 sHS[WPB][16 * 16];  // shared h3/h4/h5 buffer

    const int tid = threadIdx.x;

    // ---- Stage & swizzle W1 into LDS (once per block) ----
    for (int idx = tid; idx < 4 * 4 * 32 * 16; idx += TPB) {
        int e = idx & 15, ln = (idx >> 4) & 31, t = (idx >> 9) & 3, c = (idx >> 11) & 3;
        int k = c * 32 + (ln >> 4) * 8 + (e & 7) + (e >> 3) * 16;
        int n = t * 16 + (ln & 15);
        sW1f[idx] = (_Float16)gW1[k * 64 + n];
    }
    __syncthreads();

    const int wave = tid >> 5;
    const int lane = tid & 31;
    const int row  = lane & 15;          // A-matrix row / C & B matrix column
    const int hi   = lane >> 4;          // lane-half select
    const int kb   = hi * 8;             // A-fragment K base within a 32-chunk
    _Float16* h1 = sH1[wave];
    _Float16* h2 = sH2[wave];
    _Float16* hs = sHS[wave];

    // ---- Register-resident B fragments for W2..W6 (built once, padded) ----
    // B-fragment layout: lane L holds column n = L&15, K = (L>>4)*8 + (e&7) + (e>>3)*16
    v16h w2f[2][2], w3f, w4f, w5f, w6f;
#pragma unroll
    for (int c = 0; c < 2; ++c)
#pragma unroll
        for (int t = 0; t < 2; ++t)
#pragma unroll
            for (int e = 0; e < 16; ++e) {
                int k = c * 32 + hi * 8 + (e & 7) + (e >> 3) * 16;
                w2f[c][t][e] = (_Float16)gW2[k * 32 + t * 16 + row];
            }
#pragma unroll
    for (int e = 0; e < 16; ++e) {
        int k = hi * 8 + (e & 7) + (e >> 3) * 16;
        w3f[e] = (_Float16)gW3[k * 16 + row];
        w4f[e] = (k < 16 && row < 8) ? (_Float16)gW4[k * 8 + row] : (_Float16)0.0f;
        w5f[e] = (k < 8  && row < 4) ? (_Float16)gW5[k * 4 + row] : (_Float16)0.0f;
        w6f[e] = (k < 4  && row == 0) ? (_Float16)gW6[k]          : (_Float16)0.0f;
    }
    // ---- Per-lane register-resident biases ----
    float bias1[4], bias2[2];
#pragma unroll
    for (int t = 0; t < 4; ++t) bias1[t] = gB1[t * 16 + row];
#pragma unroll
    for (int t = 0; t < 2; ++t) bias2[t] = gB2[t * 16 + row];
    const float bias3 = gB3[row];
    const float bias4 = (row < 8) ? gB4[row] : 0.0f;
    const float bias5 = (row < 4) ? gB5[row] : 0.0f;
    const float b6v   = gB6[0];

    for (int tile = blockIdx.x * WPB + wave; tile < n_tiles; tile += gridDim.x * WPB) {
        const int tbase = tile * 16;
        int edge_raw = tbase + row;
        int edge = (edge_raw < E) ? edge_raw : (E - 1);
        const float* fi = nf + (long long)ei[edge]     * 64;
        const float* fj = nf + (long long)ei[E + edge] * 64;

        // ---- Gather endpoint features straight into layer-1 A fragments ----
        v16h a1[4];
        float dotp = 0.f, ni2 = 0.f, nj2 = 0.f;
#pragma unroll
        for (int g = 0; g < 4; ++g) {
            const float* pi = fi + g * 16 + kb;
            const float* pj = fj + g * 16 + kb;
            int c = g >> 1, h = (g & 1) * 8;
#pragma unroll
            for (int t = 0; t < 8; ++t) {
                float vi = pi[t], vj = pj[t];
                dotp += vi * vj; ni2 += vi * vi; nj2 += vj * vj;
                a1[c][h + t]     = (_Float16)vi;   // x_i -> e[0:64]  (chunks 0,1)
                a1[2 + c][h + t] = (_Float16)vj;   // x_j -> e[64:128] (chunks 2,3)
            }
        }

        // ---- Cosine similarity (f32, lane-pair reduction) ----
        float dotF = dotp + __shfl_xor(dotp, 16, 32);
        float niF  = ni2  + __shfl_xor(ni2, 16, 32);
        float njF  = nj2  + __shfl_xor(nj2, 16, 32);
        if (hi == 0 && edge_raw < E) {
            float denom = fmaxf(sqrtf(niF), 1e-8f) * fmaxf(sqrtf(njF), 1e-8f);
            out_cos[edge_raw] = dotF / denom;
        }

        // ---- Layer 1: [16x128] x [128x64] (B fragments streamed from LDS) ----
#pragma unroll
        for (int t = 0; t < 4; ++t) {
            v8f acc = {};
#pragma unroll
            for (int c = 0; c < 4; ++c) {
                v16h b = *(const v16h*)(sW1f + ((c * 4 + t) * 32 + lane) * 16);
                acc = __builtin_amdgcn_wmma_f32_16x16x32_f16(false, a1[c], false, b,
                                                             (short)0, acc, false, false);
            }
#pragma unroll
            for (int r = 0; r < 8; ++r) {
                float v = fmaxf(acc[r] + bias1[t], 0.f);
                h1[(r + 8 * hi) * 64 + t * 16 + row] = (_Float16)v;
            }
        }

        // ---- Layer 2: [16x64] x [64x32] ----
        v16h a2[2];
#pragma unroll
        for (int c = 0; c < 2; ++c) {
            const _Float16* p = h1 + row * 64 + c * 32 + kb;
#pragma unroll
            for (int t = 0; t < 8; ++t) { a2[c][t] = p[t]; a2[c][8 + t] = p[16 + t]; }
        }
#pragma unroll
        for (int t = 0; t < 2; ++t) {
            v8f acc = {};
#pragma unroll
            for (int c = 0; c < 2; ++c)
                acc = __builtin_amdgcn_wmma_f32_16x16x32_f16(false, a2[c], false, w2f[c][t],
                                                             (short)0, acc, false, false);
#pragma unroll
            for (int r = 0; r < 8; ++r) {
                float v = fmaxf(acc[r] + bias2[t], 0.f);
                h2[(r + 8 * hi) * 32 + t * 16 + row] = (_Float16)v;
            }
        }

        // ---- Layer 3: [16x32] x [32x16] ----
        {
            v16h a3;
            const _Float16* p = h2 + row * 32 + kb;
#pragma unroll
            for (int t = 0; t < 8; ++t) { a3[t] = p[t]; a3[8 + t] = p[16 + t]; }
            v8f acc = {};
            acc = __builtin_amdgcn_wmma_f32_16x16x32_f16(false, a3, false, w3f,
                                                         (short)0, acc, false, false);
#pragma unroll
            for (int r = 0; r < 8; ++r) {
                float v = fmaxf(acc[r] + bias3, 0.f);
                hs[(r + 8 * hi) * 16 + row] = (_Float16)v;
            }
        }

        // ---- Layer 4: [16x16 pad32] x [16x8 pad 32x16] ----
        {
            v16h a4 = {};
            const _Float16* p = hs + row * 16 + kb;
#pragma unroll
            for (int t = 0; t < 8; ++t) a4[t] = p[t];   // K>=16 padded with zeros
            v8f acc = {};
            acc = __builtin_amdgcn_wmma_f32_16x16x32_f16(false, a4, false, w4f,
                                                         (short)0, acc, false, false);
#pragma unroll
            for (int r = 0; r < 8; ++r) {
                float v = (row < 8) ? fmaxf(acc[r] + bias4, 0.f) : 0.f;
                hs[(r + 8 * hi) * 16 + row] = (_Float16)v;   // cols 8..15 forced to 0
            }
        }

        // ---- Layer 5: [16x8 pad32] x [8x4 pad 32x16] ----
        {
            v16h a5 = {};
            const _Float16* p = hs + row * 16 + kb;
#pragma unroll
            for (int t = 0; t < 8; ++t) a5[t] = p[t];
            v8f acc = {};
            acc = __builtin_amdgcn_wmma_f32_16x16x32_f16(false, a5, false, w5f,
                                                         (short)0, acc, false, false);
#pragma unroll
            for (int r = 0; r < 8; ++r) {
                float v = (row < 4) ? fmaxf(acc[r] + bias5, 0.f) : 0.f;
                hs[(r + 8 * hi) * 16 + row] = (_Float16)v;   // cols 4..15 forced to 0
            }
        }

        // ---- Layer 6: [16x4 pad32] x [4x1 pad 32x16] -> column 0 of C ----
        {
            v16h a6 = {};
            const _Float16* p = hs + row * 16 + kb;
#pragma unroll
            for (int t = 0; t < 8; ++t) a6[t] = p[t];
            v8f acc = {};
            acc = __builtin_amdgcn_wmma_f32_16x16x32_f16(false, a6, false, w6f,
                                                         (short)0, acc, false, false);
            if (row == 0) {     // lanes 0 (rows 0-7) and 16 (rows 8-15) hold column 0
#pragma unroll
                for (int r = 0; r < 8; ++r) {
                    int er = tbase + r + 8 * hi;
                    if (er < E) out_attr[er] = acc[r] + b6v;
                }
            }
        }
    }
}

extern "C" void kernel_launch(void* const* d_in, const int* in_sizes, int n_in,
                              void* d_out, int out_size, void* d_ws, size_t ws_size,
                              hipStream_t stream) {
    const float* nf = (const float*)d_in[0];
    const int*   ei = (const int*)d_in[1];
    const float* W1 = (const float*)d_in[2];  const float* b1 = (const float*)d_in[3];
    const float* W2 = (const float*)d_in[4];  const float* b2 = (const float*)d_in[5];
    const float* W3 = (const float*)d_in[6];  const float* b3 = (const float*)d_in[7];
    const float* W4 = (const float*)d_in[8];  const float* b4 = (const float*)d_in[9];
    const float* W5 = (const float*)d_in[10]; const float* b5 = (const float*)d_in[11];
    const float* W6 = (const float*)d_in[12]; const float* b6 = (const float*)d_in[13];

    const int E = in_sizes[1] / 2;            // edge_index is [2, E]
    const int n_tiles = (E + 15) / 16;
    int blocks = (n_tiles + WPB - 1) / WPB;
    if (blocks > 2048) blocks = 2048;         // persistent waves amortize weight staging

    float* out_attr = (float*)d_out;          // [E] edge-MLP output
    float* out_cos  = out_attr + E;           // [E] cosine similarity

    disj_gnn_fused<<<blocks, TPB, 0, stream>>>(
        nf, ei, W1, b1, W2, b2, W3, b3, W4, b4, W5, b5, W6, b6,
        out_attr, out_cos, E, n_tiles);
}
